// VectorQuantizer_24575802868281
// MI455X (gfx1250) — compile-verified, matrix operand order checked by reading, not run on hardware
//
#include <hip/hip_runtime.h>
#include <hip/hip_bf16.h>

typedef _Float16 h4_t   __attribute__((ext_vector_type(4)));
typedef _Float16 h8_t   __attribute__((ext_vector_type(8)));
typedef _Float16 v16h_t __attribute__((ext_vector_type(16)));
typedef float    v8f_t  __attribute__((ext_vector_type(8)));

#define DIM      256
#define K_EMB    4096
#define N_ROWS   65536
#define M_BLOCK  64      // rows per block (2 waves x 32 rows)
#define LDS_STRIDE 264   // 256 + 8-half pad -> conflict-free 16B fragment loads

// workspace byte offsets
#define WS_EF16   0                       // 4096*256*2  = 2 MB
#define WS_ENORM  2097152                 // 4096*4      = 16 KB
#define WS_IDX    (WS_ENORM + 16384)      // 65536*4     = 256 KB
#define WS_PART   (WS_IDX + 262144)       // 65536*4     = 256 KB

// ---- Kernel 1: codebook f32->f16 + ||e_k||^2 -------------------------------
__global__ void prep_embed(const float* __restrict__ emb,
                           _Float16* __restrict__ ef16,
                           float* __restrict__ enorm) {
  const int k = blockIdx.x, t = threadIdx.x;
  float v = emb[(size_t)k * DIM + t];
  ef16[(size_t)k * DIM + t] = (_Float16)v;
  __shared__ float red[256];
  red[t] = v * v;
  __syncthreads();
  for (int s = 128; s > 0; s >>= 1) { if (t < s) red[t] += red[t + s]; __syncthreads(); }
  if (t == 0) enorm[k] = red[0];
}

__device__ __forceinline__ v16h_t combine16(h8_t lo, h8_t hi) {
  return __builtin_shufflevector(lo, hi, 0,1,2,3,4,5,6,7,8,9,10,11,12,13,14,15);
}

// ---- Kernel 2: WMMA GEMM + fused argmax ------------------------------------
// 64 threads = 2 waves; each wave computes TWO 16-row tiles so each B fragment
// feeds 2 WMMAs (1:1 WMMA:VMEM). B is rotated through two register buffers so
// fragment i+2's load issues before fragment i's WMMAs (pipeline depth 2).
__global__ __launch_bounds__(64) void argmin_gemm(
    const float* __restrict__ x, const _Float16* __restrict__ ef16,
    const float* __restrict__ enorm, int* __restrict__ idx_out,
    float* __restrict__ idxf_out) {
  __shared__ _Float16 As[M_BLOCK * LDS_STRIDE];  // 33.8 KB
  const int tid  = threadIdx.x;
  const int row0 = blockIdx.x * M_BLOCK;

  // stage x tile (64x256) to LDS as f16
  const float4* xv = (const float4*)(x + (size_t)row0 * DIM);
#pragma unroll 4
  for (int i = 0; i < 64; ++i) {
    int e = tid + 64 * i;             // float4 index within tile
    float4 f = xv[e];
    int r = e >> 6;                   // / (256/4)
    int c = (e & 63) << 2;
    h4_t h = { (_Float16)f.x, (_Float16)f.y, (_Float16)f.z, (_Float16)f.w };
    *(h4_t*)&As[r * LDS_STRIDE + c] = h;
  }
  __syncthreads();

  const int lane  = tid & 31;
  const int wid   = tid >> 5;         // 0..1
  const int hh    = lane >> 4;        // half-wave id
  const int l16   = lane & 15;
  const int kbase = hh * 8;           // ISA 16-bit A/B lane K-offset

  // hoist both A tiles into registers: 2 x 8 x v16h = 128 VGPRs
  const _Float16* aRow0 = &As[(wid * 32 +      l16) * LDS_STRIDE];
  const _Float16* aRow1 = &As[(wid * 32 + 16 + l16) * LDS_STRIDE];
  v16h_t a0[8], a1[8];
#pragma unroll
  for (int i = 0; i < 8; ++i) {
    int d = i * 32;
    a0[i] = combine16(*(const h8_t*)(aRow0 + d + kbase),
                      *(const h8_t*)(aRow0 + d + kbase + 16));
    a1[i] = combine16(*(const h8_t*)(aRow1 + d + kbase),
                      *(const h8_t*)(aRow1 + d + kbase + 16));
  }

  float bestVal0[8], bestVal1[8]; int bestIdx0[8], bestIdx1[8];
#pragma unroll
  for (int v = 0; v < 8; ++v) {
    bestVal0[v] = -3.4e38f; bestIdx0[v] = 0;
    bestVal1[v] = -3.4e38f; bestIdx1[v] = 0;
  }

  for (int kc = 0; kc < K_EMB; kc += 16) {
    const int col = kc + l16;
    const _Float16* bRow = ef16 + (size_t)col * DIM + kbase;
    auto ldb = [&](int i) -> v16h_t {
      return combine16(*(const h8_t*)(bRow + i * 32),
                       *(const h8_t*)(bRow + i * 32 + 16));
    };
    const float en = enorm[col];       // issue early; waited at the FMA below
    v8f_t acc0 = {0.f,0.f,0.f,0.f,0.f,0.f,0.f,0.f};
    v8f_t acc1 = {0.f,0.f,0.f,0.f,0.f,0.f,0.f,0.f};
    v16h_t bA = ldb(0);
    v16h_t bB = ldb(1);
#pragma unroll
    for (int i = 0; i < 8; ++i) {
      v16h_t bc = bA;                  // fragment i (already in flight/ready)
      bA = bB;
      if (i < 6) bB = ldb(i + 2);      // issue i+2 before WMMAs of i
      acc0 = __builtin_amdgcn_wmma_f32_16x16x32_f16(false, a0[i], false, bc,
                                                    (short)0, acc0, false, false);
      acc1 = __builtin_amdgcn_wmma_f32_16x16x32_f16(false, a1[i], false, bc,
                                                    (short)0, acc1, false, false);
    }
#pragma unroll
    for (int v = 0; v < 8; ++v) {
      float s0 = 2.0f * acc0[v] - en;  // = -(d2 - ||x||^2); argmax == argmin d2
      float s1 = 2.0f * acc1[v] - en;
      if (s0 > bestVal0[v]) { bestVal0[v] = s0; bestIdx0[v] = col; }
      if (s1 > bestVal1[v]) { bestVal1[v] = s1; bestIdx1[v] = col; }
    }
  }

  // argmax reduction over the 16 lanes holding each row (tie -> smaller index)
#pragma unroll
  for (int v = 0; v < 8; ++v) {
#pragma unroll
    for (int off = 8; off >= 1; off >>= 1) {
      float ov0 = __shfl_xor(bestVal0[v], off, 16);
      int   oi0 = __shfl_xor(bestIdx0[v], off, 16);
      if (ov0 > bestVal0[v] || (ov0 == bestVal0[v] && oi0 < bestIdx0[v])) {
        bestVal0[v] = ov0; bestIdx0[v] = oi0;
      }
      float ov1 = __shfl_xor(bestVal1[v], off, 16);
      int   oi1 = __shfl_xor(bestIdx1[v], off, 16);
      if (ov1 > bestVal1[v] || (ov1 == bestVal1[v] && oi1 < bestIdx1[v])) {
        bestVal1[v] = ov1; bestIdx1[v] = oi1;
      }
    }
  }

  if (l16 == 0) {
#pragma unroll
    for (int v = 0; v < 8; ++v) {
      int r0 = row0 + wid * 32      + hh * 8 + v;  // tile 0
      int r1 = row0 + wid * 32 + 16 + hh * 8 + v;  // tile 1
      idx_out[r0]  = bestIdx0[v];
      idxf_out[r0] = (float)bestIdx0[v];
      idx_out[r1]  = bestIdx1[v];
      idxf_out[r1] = (float)bestIdx1[v];
    }
  }
}

// ---- Kernel 3: gather quantized + per-row loss partial ---------------------
__global__ void gather_loss(const float* __restrict__ x, const float* __restrict__ emb,
                            const int* __restrict__ idx, float* __restrict__ qout,
                            float* __restrict__ partial) {
  const int n = blockIdx.x, t = threadIdx.x;
  const int k = idx[n];
  float e  = emb[(size_t)k * DIM + t];
  float xv = x[(size_t)n * DIM + t];
  qout[(size_t)n * DIM + t] = e;       // straight-through value == quantized
  float d = e - xv;
  __shared__ float red[256];
  red[t] = d * d;
  __syncthreads();
  for (int s = 128; s > 0; s >>= 1) { if (t < s) red[t] += red[t + s]; __syncthreads(); }
  if (t == 0) partial[n] = red[0];
}

// ---- Kernel 4: deterministic final loss reduction --------------------------
__global__ void loss_reduce(const float* __restrict__ partial, float* __restrict__ loss) {
  const int t = threadIdx.x;
  float s = 0.f;
  for (int i = 0; i < 256; ++i) s += partial[t + 256 * i];
  __shared__ float red[256];
  red[t] = s;
  __syncthreads();
  for (int w = 128; w > 0; w >>= 1) { if (t < w) red[t] += red[t + w]; __syncthreads(); }
  if (t == 0) loss[0] = red[0] * (1.25f / 16777216.0f);  // (1 + 0.25) * mean
}

extern "C" void kernel_launch(void* const* d_in, const int* in_sizes, int n_in,
                              void* d_out, int out_size, void* d_ws, size_t ws_size,
                              hipStream_t stream) {
  const float* x   = (const float*)d_in[0];   // [16,4096,256] f32
  const float* emb = (const float*)d_in[1];   // [4096,256] f32

  float* out  = (float*)d_out;
  float* qout = out;                          // 16,777,216 floats
  float* idxf = out + 16777216;               // 65,536 floats (indices)
  float* loss = out + 16777216 + 65536;       // 1 float

  char* ws = (char*)d_ws;
  _Float16* ef16  = (_Float16*)(ws + WS_EF16);
  float*    enorm = (float*)(ws + WS_ENORM);
  int*      idx   = (int*)(ws + WS_IDX);
  float*    part  = (float*)(ws + WS_PART);

  prep_embed<<<K_EMB, 256, 0, stream>>>(emb, ef16, enorm);
  argmin_gemm<<<N_ROWS / M_BLOCK, 64, 0, stream>>>(x, ef16, enorm, idx, idxf);
  gather_loss<<<N_ROWS, 256, 0, stream>>>(x, emb, idx, qout, part);
  loss_reduce<<<1, 256, 0, stream>>>(part, loss);
}
